// MultiHeadAttention_26018911879696
// MI455X (gfx1250) — compile-verified, matrix operand order checked by reading, not run on hardware
//
#include <hip/hip_runtime.h>

// ---------------------------------------------------------------------------
// MI455X (gfx1250) multi-head attention forward.
// - All matmuls on v_wmma_f32_16x16x32_f16 (wave32), fp32 accumulate.
// - GEMMs use CDNA5 async global->LDS DMA (global_load_async_to_lds_b128,
//   ASYNCcnt) with double-buffered LDS tiles.
// - Attention computed transposed (S^T = K*Q^T, O^T = V^T*P^T) so softmax
//   reductions are in-lane and P^T re-enters WMMA directly from registers.
// ---------------------------------------------------------------------------

typedef __attribute__((ext_vector_type(16))) _Float16 v16h;
typedef __attribute__((ext_vector_type(8)))  _Float16 v8h;
typedef __attribute__((ext_vector_type(8)))  float    v8f;

#define WMMA_F16(a, b, c) \
    __builtin_amdgcn_wmma_f32_16x16x32_f16(false, (a), false, (b), (short)0, (c), false, false)

__device__ __forceinline__ v16h cat8(v8h lo, v8h hi) {
    return __builtin_shufflevector(lo, hi, 0,1,2,3,4,5,6,7,8,9,10,11,12,13,14,15);
}

// Low 32 bits of a flat pointer to LDS == byte offset in LDS (aperture rule).
__device__ __forceinline__ unsigned lds_off(const void* p) {
    return (unsigned)(unsigned long long)p;
}

// CDNA5 async DMA: global -> LDS, 16 bytes per lane, tracked by ASYNCcnt.
__device__ __forceinline__ void async_copy_b128(unsigned ldsOff, const void* gaddr) {
    asm volatile("global_load_async_to_lds_b128 %0, %1, off"
                 :: "v"(ldsOff), "v"(gaddr)
                 : "memory");
}
__device__ __forceinline__ void wait_async0() {
    asm volatile("s_wait_asynccnt 0" ::: "memory");
}

// ---------------------------------------------------------------------------
// Kernel 1: convert X -> f16, pack Wq|Wk|Wv transposed [3072][1024],
//           W0 transposed [1024][1024].
// ---------------------------------------------------------------------------
__global__ __launch_bounds__(256) void convert_kernel(
    const float* __restrict__ X,  const float* __restrict__ Wq,
    const float* __restrict__ Wk, const float* __restrict__ Wv,
    const float* __restrict__ W0,
    _Float16* __restrict__ Xh, _Float16* __restrict__ Wqkvt,
    _Float16* __restrict__ W0t)
{
    const long i = (long)blockIdx.x * 256 + threadIdx.x;
    const long NX = 4096L * 1024;   // X elements
    const long NW = 3072L * 1024;   // packed qkv weights
    const long NO = 1024L * 1024;   // W0
    if (i < NX) {
        Xh[i] = (_Float16)X[i];
    } else if (i < NX + NW) {
        const long j = i - NX;
        const int n = (int)(j >> 10), k = (int)(j & 1023);
        float v;
        if      (n < 1024) v = Wq[(long)k * 1024 + n];
        else if (n < 2048) v = Wk[(long)k * 1024 + (n - 1024)];
        else               v = Wv[(long)k * 1024 + (n - 2048)];
        Wqkvt[j] = (_Float16)v;
    } else if (i < NX + NW + NO) {
        const long j = i - NX - NW;
        const int n = (int)(j >> 10), k = (int)(j & 1023);
        W0t[j] = (_Float16)W0[(long)k * 1024 + n];
    }
}

// ---------------------------------------------------------------------------
// Generic f16 WMMA GEMM:  C[M x N] = A[M x 1024] * Bt[N x 1024]^T
// Block: 256 threads = 8 waves; block tile 128(M) x 64(N); wave tile 32x32.
// K-step 64, double-buffered LDS filled by async global->LDS DMA.
// MODE 0: QKV epilogue (scatter f16 into Q[B,H,S,d], K[B,H,S,d], Vt[B,H,d,S])
// MODE 1: plain f32 row-major store.
// ---------------------------------------------------------------------------
template <int MODE>
__global__ __launch_bounds__(256) void gemm_f16_kernel(
    const _Float16* __restrict__ A, const _Float16* __restrict__ Bt,
    _Float16* __restrict__ oQ, _Float16* __restrict__ oK,
    _Float16* __restrict__ oVt, float* __restrict__ oF)
{
    constexpr int Kd = 1024;
    constexpr int KSTEP = 64;
    // Row stride 72 halves = 144 B (16B-aligned rows, padded vs bank conflicts)
    __shared__ _Float16 As[2][128][72];   // 36864 B
    __shared__ _Float16 Bs[2][64][72];    // 18432 B

    const int tid  = threadIdx.x;
    const int lane = tid & 31;
    const int w    = tid >> 5;
    const int half = lane >> 4;
    const int l16  = lane & 15;
    const int wm   = (w >> 1) * 32;    // wave M offset in block tile (0..96)
    const int wn   = (w & 1)  * 32;    // wave N offset (0 or 32)
    const long blockM = (long)blockIdx.x * 128;
    const long blockN = (long)blockIdx.y * 64;

    v8f acc[2][2] = {};

    // Async tile loader: 256 threads; 16B per lane per issue.
    const int ar = tid >> 3;           // 0..31
    const int ac = (tid & 7) * 8;      // 0,8,..,56
    auto load_tile = [&](int buf, int k0) {
#pragma unroll
        for (int p = 0; p < 4; ++p)
            async_copy_b128(lds_off(&As[buf][p * 32 + ar][ac]),
                            &A[(blockM + p * 32 + ar) * Kd + k0 + ac]);
#pragma unroll
        for (int p = 0; p < 2; ++p)
            async_copy_b128(lds_off(&Bs[buf][p * 32 + ar][ac]),
                            &Bt[(blockN + p * 32 + ar) * Kd + k0 + ac]);
    };

    load_tile(0, 0);
    wait_async0();
    __syncthreads();

    for (int k0 = 0; k0 < Kd; k0 += KSTEP) {
        const int cur = (k0 >> 6) & 1;
        const bool more = (k0 + KSTEP) < Kd;
        if (more) load_tile(cur ^ 1, k0 + KSTEP);   // DMA next tile in flight

        // Compute 2 K=32 chunks from the current buffer: 8 WMMAs.
#pragma unroll
        for (int c = 0; c < 2; ++c) {
            v16h a[2], b[2];
#pragma unroll
            for (int ms = 0; ms < 2; ++ms) {
                const _Float16* src = &As[cur][wm + ms * 16 + l16][c * 32];
                a[ms] = cat8(*(const v8h*)&src[8 * half],
                             *(const v8h*)&src[16 + 8 * half]);
            }
#pragma unroll
            for (int ns = 0; ns < 2; ++ns) {
                const _Float16* src = &Bs[cur][wn + ns * 16 + l16][c * 32];
                b[ns] = cat8(*(const v8h*)&src[8 * half],
                             *(const v8h*)&src[16 + 8 * half]);
            }
#pragma unroll
            for (int ms = 0; ms < 2; ++ms)
#pragma unroll
                for (int ns = 0; ns < 2; ++ns)
                    acc[ms][ns] = WMMA_F16(a[ms], b[ns], acc[ms][ns]);
        }

        if (more) {
            wait_async0();      // next buffer fully in LDS (this wave's DMAs)
            __syncthreads();    // ... and everyone else's; also fences reads
        }
    }

#pragma unroll
    for (int ms = 0; ms < 2; ++ms)
#pragma unroll
        for (int ns = 0; ns < 2; ++ns)
#pragma unroll
            for (int r = 0; r < 8; ++r) {
                const long gm = blockM + wm + ms * 16 + r + 8 * half;
                const long gn = blockN + wn + ns * 16 + l16;
                const float val = acc[ms][ns][r];
                if constexpr (MODE == 0) {
                    const int bb = (int)(gm >> 11), s = (int)(gm & 2047);
                    if (gn < 1024) {
                        const int h = (int)(gn >> 6), dn = (int)(gn & 63);
                        oQ[(((long)(bb * 16 + h)) * 2048 + s) * 64 + dn] = (_Float16)val;
                    } else if (gn < 2048) {
                        const long g = gn - 1024;
                        const int h = (int)(g >> 6), dn = (int)(g & 63);
                        oK[(((long)(bb * 16 + h)) * 2048 + s) * 64 + dn] = (_Float16)val;
                    } else {
                        const long g = gn - 2048;
                        const int h = (int)(g >> 6), dn = (int)(g & 63);
                        oVt[(((long)(bb * 16 + h)) * 64 + dn) * 2048 + s] = (_Float16)val;
                    }
                } else {
                    oF[gm * 1024 + gn] = val;
                }
            }
}

// ---------------------------------------------------------------------------
// Kernel 3: flash attention, fully transposed. Grid: (S/128, B*H).
// 8 waves; wave owns 16 q rows (one q per lane-mod-16 column).
//   S^T = K * Q^T      (A = K frags, B = Q frags)
//   O^T = V^T * P^T    (A = V^T frags, B = P^T built in-registers)
// Q[B,H,S,64], K[B,H,S,64] row-major f16; V transposed [B,H,64,S] f16.
// Output: concat layout [B*S][1024] f16, packed 16B stores.
// ---------------------------------------------------------------------------
__global__ __launch_bounds__(256) void attn_kernel(
    const _Float16* __restrict__ Qh, const _Float16* __restrict__ Kh,
    const _Float16* __restrict__ Vt, _Float16* __restrict__ AOut)
{
    const int tid  = threadIdx.x;
    const int lane = tid & 31;
    const int w    = tid >> 5;
    const int half = lane >> 4;
    const int l16  = lane & 15;
    const int bh   = blockIdx.y;             // 0..31
    const int b    = bh >> 4, h = bh & 15;
    const int qloc = blockIdx.x * 128 + w * 16;
    const long qkBase = (long)bh * 2048 * 64;   // Q/K base for this (b,h)
    const long vBase  = (long)bh * 64 * 2048;   // Vt base

    // Q^T B-fragments (d = 64 -> two K=32 chunks); lane = q column qloc + l16.
    v16h qf[2];
    {
        const _Float16* qrow = Qh + qkBase + (long)(qloc + l16) * 64;
#pragma unroll
        for (int c = 0; c < 2; ++c)
            qf[c] = cat8(*(const v8h*)&qrow[c * 32 + 8 * half],
                         *(const v8h*)&qrow[c * 32 + 16 + 8 * half]);
    }

    // Per-lane softmax state for this lane's q column.
    float m_run = -1e30f;
    float l_run = 0.0f;
    v8f o[4] = {};                            // O^T accumulators: dn rows x q col

    const float sc    = 0.125f;               // 1/sqrt(64)
    const float LOG2E = 1.44269504088896340736f;

    for (int kt = 0; kt < 2048; kt += 32) {
        // K A-fragments: lane = key row, contiguous d runs.
        v16h kf[2][2];
#pragma unroll
        for (int ms = 0; ms < 2; ++ms) {
            const _Float16* krow = Kh + qkBase + (long)(kt + ms * 16 + l16) * 64;
#pragma unroll
            for (int c = 0; c < 2; ++c)
                kf[ms][c] = cat8(*(const v8h*)&krow[c * 32 + 8 * half],
                                 *(const v8h*)&krow[c * 32 + 16 + 8 * half]);
        }
        // S^T tiles: rows = keys (ms*16 + r + 8*half), col = q (l16).
        v8f s[2] = {};
#pragma unroll
        for (int ms = 0; ms < 2; ++ms) {
            s[ms] = WMMA_F16(kf[ms][0], qf[0], s[ms]);
            s[ms] = WMMA_F16(kf[ms][1], qf[1], s[ms]);
        }

        // Online softmax over the 32 keys of this tile (16 in-lane, 16 in
        // lane^16): in-lane reduce + one shfl_xor(16).
        float tmax = -1e30f;
#pragma unroll
        for (int r = 0; r < 8; ++r) {
            s[0][r] *= sc; s[1][r] *= sc;
            tmax = fmaxf(tmax, fmaxf(s[0][r], s[1][r]));
        }
        tmax = fmaxf(tmax, __shfl_xor(tmax, 16, 32));
        const float mnew  = fmaxf(m_run, tmax);
        const float alpha = exp2f((m_run - mnew) * LOG2E);

        float p0[8], p1[8], psum = 0.0f;
#pragma unroll
        for (int r = 0; r < 8; ++r) {
            p0[r] = exp2f((s[0][r] - mnew) * LOG2E);
            p1[r] = exp2f((s[1][r] - mnew) * LOG2E);
            psum += p0[r] + p1[r];
        }
        psum += __shfl_xor(psum, 16, 32);
        l_run = l_run * alpha + psum;
        m_run = mnew;
#pragma unroll
        for (int nf = 0; nf < 4; ++nf)
#pragma unroll
            for (int r = 0; r < 8; ++r)
                o[nf][r] *= alpha;

        // P^T B-fragment directly from registers: element e<8 needs key
        // 8*half+e  (= s[0] vgpr e); e>=8 needs key 16+8*half+(e-8) (= s[1]).
        v16h pb;
#pragma unroll
        for (int e = 0; e < 8; ++e) {
            pb[e]     = (_Float16)p0[e];
            pb[e + 8] = (_Float16)p1[e];
        }

        // V^T A-fragments: lane = dn row, contiguous key runs.
#pragma unroll
        for (int nf = 0; nf < 4; ++nf) {
            const _Float16* vrow = Vt + vBase + (long)(nf * 16 + l16) * 2048 + kt;
            v16h vf = cat8(*(const v8h*)&vrow[8 * half], *(const v8h*)&vrow[16 + 8 * half]);
            o[nf] = WMMA_F16(vf, pb, o[nf]);
        }
    }

    // Normalize + store. Lane's q row is fixed; dn runs are contiguous ->
    // pack 8 halves and do 16B stores.
    const float inv_l = 1.0f / l_run;
    const long row = (long)b * 2048 + qloc + l16;
    _Float16* dst = AOut + row * 1024 + h * 64;
#pragma unroll
    for (int nf = 0; nf < 4; ++nf) {
        v8h ph;
#pragma unroll
        for (int r = 0; r < 8; ++r)
            ph[r] = (_Float16)(o[nf][r] * inv_l);
        *(v8h*)&dst[nf * 16 + 8 * half] = ph;
    }
}

// ---------------------------------------------------------------------------
// Host launcher
// ---------------------------------------------------------------------------
extern "C" void kernel_launch(void* const* d_in, const int* in_sizes, int n_in,
                              void* d_out, int out_size, void* d_ws, size_t ws_size,
                              hipStream_t stream) {
    (void)in_sizes; (void)n_in; (void)out_size; (void)ws_size;
    const float* X  = (const float*)d_in[0];
    const float* Wq = (const float*)d_in[1];
    const float* Wk = (const float*)d_in[2];
    const float* Wv = (const float*)d_in[3];
    const float* W0 = (const float*)d_in[4];
    float* out = (float*)d_out;

    char* p = (char*)d_ws;
    auto take = [&](size_t bytes) { char* q = p; p += bytes; return q; };
    _Float16* Xh    = (_Float16*)take(4096L * 1024 * 2);  //  8 MB
    _Float16* Wqkvt = (_Float16*)take(3072L * 1024 * 2);  //  6 MB
    _Float16* W0t   = (_Float16*)take(1024L * 1024 * 2);  //  2 MB
    _Float16* Qh    = (_Float16*)take(4096L * 1024 * 2);  //  8 MB
    _Float16* Kh    = (_Float16*)take(4096L * 1024 * 2);  //  8 MB
    _Float16* Vt    = (_Float16*)take(4096L * 1024 * 2);  //  8 MB
    _Float16* AOut  = (_Float16*)take(4096L * 1024 * 2);  //  8 MB  (48 MB total)

    // 1) convert / pack
    convert_kernel<<<(8 * 1024 * 1024) / 256, 256, 0, stream>>>(
        X, Wq, Wk, Wv, W0, Xh, Wqkvt, W0t);
    // 2) fused QKV projection: [4096x1024] x [1024x3072]
    gemm_f16_kernel<0><<<dim3(32, 48), 256, 0, stream>>>(
        Xh, Wqkvt, Qh, Kh, Vt, nullptr);
    // 3) flash attention per (b,h, q-tile of 128)
    attn_kernel<<<dim3(16, 32), 256, 0, stream>>>(Qh, Kh, Vt, AOut);
    // 4) output projection: [4096x1024] x [1024x1024] -> f32
    gemm_f16_kernel<1><<<dim3(32, 16), 256, 0, stream>>>(
        AOut, W0t, nullptr, nullptr, nullptr, out);
}
